// ODEDecoder2_40106404610782
// MI455X (gfx1250) — compile-verified, matrix operand order checked by reading, not run on hardware
//
#include <hip/hip_runtime.h>
#include <hip/hip_bf16.h>

typedef __attribute__((ext_vector_type(8)))  float  v8f;
typedef __attribute__((ext_vector_type(16))) __bf16 v16bf;

#define BB 32
#define NN 64
#define EE 4032
#define TT 11
#define FF 4

__device__ __forceinline__ unsigned short f32_to_bf16(float f) {
  unsigned int u = __float_as_uint(f);
  u += 0x7FFFu + ((u >> 16) & 1u);            // round-to-nearest-even
  return (unsigned short)(u >> 16);
}
__device__ __forceinline__ float bf16_to_f32(unsigned short h) {
  return __uint_as_float((unsigned int)h << 16);
}
// ISA 16-bit fragment position: kk (0..31) -> lane-half g, packed slot p
__device__ __forceinline__ void kpos(int kk, int& g, int& p) {
  if (kk < 16) { g = kk >> 3; p = kk & 7; }
  else         { int k2 = kk - 16; g = k2 >> 3; p = 8 + (k2 & 7); }
}

// ---------------------------------------------------------------------------
// Pack W2 (2,256,256) f32 -> bf16 B-fragment layout (k, nt, ks, lane, p)
// ---------------------------------------------------------------------------
__global__ __launch_bounds__(256) void pack_w2_kernel(
    const float* __restrict__ W2, unsigned short* __restrict__ w2f) {
  int tid = blockIdx.x * 256 + threadIdx.x;   // 0 .. 131071
  int k = tid >> 16, rem = tid & 65535, h = rem >> 8, o = rem & 255;
  int nt = o >> 4, col = o & 15, ks = h >> 5, g, p;
  kpos(h & 31, g, p);
  int lane = g * 16 + col;
  w2f[(((size_t)(k * 16 + nt) * 8 + ks) * 32 + lane) * 16 + p] = f32_to_bf16(W2[tid]);
}

// Pack Wo1 (260,256) -> bf16 frags, K zero-padded to 288 (9 ks-steps)
__global__ __launch_bounds__(256) void pack_wo1_kernel(
    const float* __restrict__ Wo1, unsigned short* __restrict__ wo1f) {
  int tid = blockIdx.x * 256 + threadIdx.x;   // 0 .. 73727 (16*9*32*16)
  int p = tid & 15, lane = (tid >> 4) & 31;
  int ks = (tid >> 9) % 9, nt = tid / (9 * 512);
  int g = lane >> 4, col = lane & 15;
  int kk = (p < 8) ? (g * 8 + p) : (16 + g * 8 + (p - 8));
  int f = ks * 32 + kk, o = nt * 16 + col;
  wo1f[tid] = (f < 260) ? f32_to_bf16(Wo1[f * 256 + o]) : (unsigned short)0;
}

// Pack Wo2 (256,256) -> bf16 frags (8 ks-steps)
__global__ __launch_bounds__(256) void pack_wo2_kernel(
    const float* __restrict__ Wo2, unsigned short* __restrict__ wo2f) {
  int tid = blockIdx.x * 256 + threadIdx.x;   // 0 .. 65535
  int p = tid & 15, lane = (tid >> 4) & 31;
  int ks = (tid >> 9) & 7, nt = tid >> 12;
  int g = lane >> 4, col = lane & 15;
  int kk = (p < 8) ? (g * 8 + p) : (16 + g * 8 + (p - 8));
  wo2f[tid] = f32_to_bf16(Wo2[(ks * 32 + kk) * 256 + nt * 16 + col]);
}

// x0[b][n][f] = inputs[b][n][0][f]
__global__ __launch_bounds__(256) void init_x0_kernel(
    const float* __restrict__ inputs, float* __restrict__ x0) {
  int tid = blockIdx.x * 256 + threadIdx.x;   // 0 .. 8191
  x0[tid] = inputs[(tid >> 2) * (TT * FF) + (tid & 3)];
}

// ---------------------------------------------------------------------------
// Edge kernel: one WG per (b, receiver i); M=64 senders (diagonal weighted 0).
// ---------------------------------------------------------------------------
__global__ __launch_bounds__(256) void edge_kernel(
    const float* __restrict__ x, const float* __restrict__ edges,
    const float* __restrict__ W1, const float* __restrict__ b1,
    const float* __restrict__ b2, const unsigned short* __restrict__ w2f,
    float* __restrict__ agg)
{
  const int b = blockIdx.x >> 6;
  const int i = blockIdx.x & 63;
  const int tid = threadIdx.x;

  __shared__ __align__(32) unsigned short hfrag[4 * 8 * 32 * 16]; // 32 KB
  __shared__ float pre[64][8];
  __shared__ float coef[2][64];
  __shared__ float aggpart[256];

  {
    int j = tid >> 2, f = tid & 3;
    pre[j][f]     = x[(b * NN + j) * FF + f];
    pre[j][4 + f] = x[(b * NN + i) * FF + f];
  }
  if (tid < 128) {
    int j = tid >> 1, k = tid & 1;
    float c = 0.f;
    if (j != i) {
      int e = i * 63 + (j < i ? j : j - 1);
      c = edges[((size_t)b * EE + e) * 2 + k];
    }
    coef[k][j] = c;
  }
  aggpart[tid] = 0.f;
  __syncthreads();

  const int lane   = tid & 31;
  const int wave   = tid >> 5;
  const int mt     = wave & 3;
  const int ntbase = (wave >> 2) * 8;

  for (int k = 0; k < 2; ++k) {
    // ---- GEMM1 (64x8 @ 8x256) + relu + pack bf16 A-fragments ----
    {
      int c = (tid & 127) * 2;
      int mlo = (tid >> 7) * 32;
      float wA[8], wB[8];
      #pragma unroll
      for (int f = 0; f < 8; ++f) {
        wA[f] = W1[(k * 8 + f) * 256 + c];
        wB[f] = W1[(k * 8 + f) * 256 + c + 1];
      }
      float bb0 = b1[k * 256 + c], bb1 = b1[k * 256 + c + 1];
      int ks = c >> 5, g, p;
      kpos(c & 31, g, p);
      for (int m = mlo; m < mlo + 32; ++m) {
        float s0 = bb0, s1 = bb1;
        #pragma unroll
        for (int f = 0; f < 8; ++f) {
          float pv = pre[m][f];
          s0 = fmaf(pv, wA[f], s0);
          s1 = fmaf(pv, wB[f], s1);
        }
        s0 = fmaxf(s0, 0.f); s1 = fmaxf(s1, 0.f);
        unsigned int packed =
            (unsigned int)f32_to_bf16(s0) | ((unsigned int)f32_to_bf16(s1) << 16);
        int idx = (((m >> 4) * 8 + ks) * 32 + (g * 16 + (m & 15))) * 16 + p;
        *(unsigned int*)&hfrag[idx] = packed;
      }
    }
    __syncthreads();

    // ---- GEMM2 (64x256 @ 256x256): batch 8 B-loads per K-step, then 8 WMMA ----
    const unsigned short* w2base = w2f + (size_t)k * 16 * 8 * 32 * 16;
    v8f acc[8];
    {
      v16bf a = *(const v16bf*)&hfrag[((mt * 8 + 0) * 32 + lane) * 16];
      v16bf bfr[8];
      #pragma unroll
      for (int nt = 0; nt < 8; ++nt)
        bfr[nt] = *(const v16bf*)
            (w2base + ((((size_t)(ntbase + nt) * 8 + 0) * 32 + lane) * 16));
      v8f z = {};
      #pragma unroll
      for (int nt = 0; nt < 8; ++nt)
        acc[nt] = __builtin_amdgcn_wmma_f32_16x16x32_bf16(
            false, a, false, bfr[nt], (short)0, z, false, false);
    }
    #pragma unroll
    for (int ks = 1; ks < 8; ++ks) {
      v16bf a = *(const v16bf*)&hfrag[((mt * 8 + ks) * 32 + lane) * 16];
      v16bf bfr[8];
      #pragma unroll
      for (int nt = 0; nt < 8; ++nt)
        bfr[nt] = *(const v16bf*)
            (w2base + ((((size_t)(ntbase + nt) * 8 + ks) * 32 + lane) * 16));
      #pragma unroll
      for (int nt = 0; nt < 8; ++nt)
        acc[nt] = __builtin_amdgcn_wmma_f32_16x16x32_bf16(
            false, a, false, bfr[nt], (short)0, acc[nt], false, false);
    }

    // ---- bias + relu + edge coefficient + column partial sums ----
    #pragma unroll
    for (int nt = 0; nt < 8; ++nt) {
      float bv = b2[k * 256 + (ntbase + nt) * 16 + (lane & 15)];
      float colsum = 0.f;
      #pragma unroll
      for (int r = 0; r < 8; ++r) {
        int m = mt * 16 + r + ((lane >> 4) << 3);
        colsum += fmaxf(acc[nt][r] + bv, 0.f) * coef[k][m];
      }
      atomicAdd(&aggpart[(ntbase + nt) * 16 + (lane & 15)], colsum);
    }
    __syncthreads();
  }

  agg[(size_t)blockIdx.x * 256 + tid] = aggpart[tid];
}

// ---------------------------------------------------------------------------
// Node kernel: 64 WGs x 32 rows. Both 256-wide layers via WMMA (K padded 288
// for layer 1). Output layer (256->4), fx = x_in + p, and RK4 stage in VALU.
// ---------------------------------------------------------------------------
__global__ __launch_bounds__(256) void node_kernel(
    const float* __restrict__ x_in, const float* __restrict__ agg,
    const unsigned short* __restrict__ wo1f, const float* __restrict__ bo1,
    const unsigned short* __restrict__ wo2f, const float* __restrict__ bo2,
    const float* __restrict__ Wo3, const float* __restrict__ bo3,
    float* __restrict__ x0, float* __restrict__ xeval,
    float* __restrict__ accum, float* __restrict__ out,
    const float* __restrict__ ts, const int* __restrict__ scale,
    int stage, int step)
{
  const int rowbase = blockIdx.x * 32;     // 64 WGs
  const int tid  = threadIdx.x;
  const int lane = tid & 31, wave = tid >> 5;
  const int mt = wave & 1;                 // 2 M-tiles
  const int ntbase = (wave >> 1) * 4;      // 4 N-tiles per wave

  __shared__ __align__(32) unsigned short afrag[2 * 9 * 32 * 16]; // 18 KB
  __shared__ __align__(4)  unsigned short p2bf[32 * 256];         // 16 KB

  // ---- build aug = [x | agg | 0-pad] as bf16 A-fragments (K=288) ----
  for (int idx = tid; idx < 32 * 144; idx += 256) {
    int m = idx / 144;
    int c = (idx % 144) * 2;
    int bn = rowbase + m;
    float v0 = (c < 4)     ? x_in[bn * 4 + c]
             : (c < 260)   ? agg[(size_t)bn * 256 + (c - 4)] : 0.f;
    int c1 = c + 1;
    float v1 = (c1 < 4)    ? x_in[bn * 4 + c1]
             : (c1 < 260)  ? agg[(size_t)bn * 256 + (c1 - 4)] : 0.f;
    int ks = c >> 5, g, p;
    kpos(c & 31, g, p);
    unsigned int packed =
        (unsigned int)f32_to_bf16(v0) | ((unsigned int)f32_to_bf16(v1) << 16);
    *(unsigned int*)&afrag[(((m >> 4) * 9 + ks) * 32 + (g * 16 + (m & 15))) * 16 + p] = packed;
  }
  __syncthreads();

  // ---- L1: p1 = relu(aug @ Wo1 + bo1), M=32 N=256 K=288 ----
  v8f acc[4];
  {
    v16bf a = *(const v16bf*)&afrag[((mt * 9 + 0) * 32 + lane) * 16];
    v16bf bfr[4];
    #pragma unroll
    for (int nt = 0; nt < 4; ++nt)
      bfr[nt] = *(const v16bf*)(wo1f + ((((size_t)(ntbase + nt) * 9 + 0) * 32 + lane) * 16));
    v8f z = {};
    #pragma unroll
    for (int nt = 0; nt < 4; ++nt)
      acc[nt] = __builtin_amdgcn_wmma_f32_16x16x32_bf16(
          false, a, false, bfr[nt], (short)0, z, false, false);
  }
  #pragma unroll
  for (int ks = 1; ks < 9; ++ks) {
    v16bf a = *(const v16bf*)&afrag[((mt * 9 + ks) * 32 + lane) * 16];
    v16bf bfr[4];
    #pragma unroll
    for (int nt = 0; nt < 4; ++nt)
      bfr[nt] = *(const v16bf*)(wo1f + ((((size_t)(ntbase + nt) * 9 + ks) * 32 + lane) * 16));
    #pragma unroll
    for (int nt = 0; nt < 4; ++nt)
      acc[nt] = __builtin_amdgcn_wmma_f32_16x16x32_bf16(
          false, a, false, bfr[nt], (short)0, acc[nt], false, false);
  }
  __syncthreads();   // afrag reads done everywhere

  // ---- bias+relu, repack p1 as A-fragments (K=256, 8 ks-steps) ----
  #pragma unroll
  for (int nt = 0; nt < 4; ++nt) {
    float bv = bo1[(ntbase + nt) * 16 + (lane & 15)];
    int c = (ntbase + nt) * 16 + (lane & 15);
    int ks = c >> 5, g, p;
    kpos(c & 31, g, p);
    #pragma unroll
    for (int r = 0; r < 8; ++r) {
      int m = mt * 16 + r + ((lane >> 4) << 3);
      float v = fmaxf(acc[nt][r] + bv, 0.f);
      afrag[(((m >> 4) * 8 + ks) * 32 + (g * 16 + (m & 15))) * 16 + p] = f32_to_bf16(v);
    }
  }
  __syncthreads();

  // ---- L2: p2 = relu(p1 @ Wo2 + bo2), M=32 N=256 K=256 ----
  v8f acc2[4];
  {
    v16bf a = *(const v16bf*)&afrag[((mt * 8 + 0) * 32 + lane) * 16];
    v16bf bfr[4];
    #pragma unroll
    for (int nt = 0; nt < 4; ++nt)
      bfr[nt] = *(const v16bf*)(wo2f + ((((size_t)(ntbase + nt) * 8 + 0) * 32 + lane) * 16));
    v8f z = {};
    #pragma unroll
    for (int nt = 0; nt < 4; ++nt)
      acc2[nt] = __builtin_amdgcn_wmma_f32_16x16x32_bf16(
          false, a, false, bfr[nt], (short)0, z, false, false);
  }
  #pragma unroll
  for (int ks = 1; ks < 8; ++ks) {
    v16bf a = *(const v16bf*)&afrag[((mt * 8 + ks) * 32 + lane) * 16];
    v16bf bfr[4];
    #pragma unroll
    for (int nt = 0; nt < 4; ++nt)
      bfr[nt] = *(const v16bf*)(wo2f + ((((size_t)(ntbase + nt) * 8 + ks) * 32 + lane) * 16));
    #pragma unroll
    for (int nt = 0; nt < 4; ++nt)
      acc2[nt] = __builtin_amdgcn_wmma_f32_16x16x32_bf16(
          false, a, false, bfr[nt], (short)0, acc2[nt], false, false);
  }

  // ---- bias+relu -> p2 (bf16, plain [m][c]) ----
  #pragma unroll
  for (int nt = 0; nt < 4; ++nt) {
    float bv = bo2[(ntbase + nt) * 16 + (lane & 15)];
    int c = (ntbase + nt) * 16 + (lane & 15);
    #pragma unroll
    for (int r = 0; r < 8; ++r) {
      int m = mt * 16 + r + ((lane >> 4) << 3);
      p2bf[m * 256 + c] = f32_to_bf16(fmaxf(acc2[nt][r] + bv, 0.f));
    }
  }
  __syncthreads();

  // ---- output layer (256->4), fx = x_in + p, RK4 stage ----
  if (tid < 128) {
    int m = tid >> 2, o = tid & 3;
    int bn = rowbase + m;
    float s = bo3[o];
    for (int c = 0; c < 256; ++c)
      s = fmaf(bf16_to_f32(p2bf[m * 256 + c]), Wo3[c * 4 + o], s);
    float kv = x_in[bn * 4 + o] + s;         // f(x) = x + p
    float dt = (ts[step + 1] - ts[step]) / (float)scale[0];
    int idx = bn * 4 + o;
    if (stage == 0) {
      float xv0 = x0[idx];
      xeval[idx] = xv0 + 0.5f * dt * kv;
      accum[idx] = xv0 + (dt / 6.f) * kv;
    } else if (stage == 1) {
      float xv0 = x0[idx];
      xeval[idx] = xv0 + 0.5f * dt * kv;
      accum[idx] += (dt / 3.f) * kv;
    } else if (stage == 2) {
      float xv0 = x0[idx];
      xeval[idx] = xv0 + dt * kv;
      accum[idx] += (dt / 3.f) * kv;
    } else {
      float xn = accum[idx] + (dt / 6.f) * kv;
      out[(size_t)bn * 40 + step * 4 + o] = xn;   // (B,N,10,F)
      x0[idx] = xn;
    }
  }
}

// ---------------------------------------------------------------------------
extern "C" void kernel_launch(void* const* d_in, const int* in_sizes, int n_in,
                              void* d_out, int out_size, void* d_ws, size_t ws_size,
                              hipStream_t stream) {
  const float* inputs = (const float*)d_in[0];
  const float* edges  = (const float*)d_in[1];
  const float* W1     = (const float*)d_in[4];
  const float* b1     = (const float*)d_in[5];
  const float* W2     = (const float*)d_in[6];
  const float* b2     = (const float*)d_in[7];
  const float* Wo1    = (const float*)d_in[8];
  const float* bo1    = (const float*)d_in[9];
  const float* Wo2    = (const float*)d_in[10];
  const float* bo2    = (const float*)d_in[11];
  const float* Wo3    = (const float*)d_in[12];
  const float* bo3    = (const float*)d_in[13];
  const float* ts     = (const float*)d_in[14];
  const int*   scale  = (const int*)d_in[16];
  float* out = (float*)d_out;

  // workspace: [w2f 131072 u16][wo1f 73728 u16][wo2f 65536 u16]
  //            [agg 524288 f32][x0 8192][xeval 8192][accum 8192]
  unsigned short* w2f  = (unsigned short*)d_ws;
  unsigned short* wo1f = w2f + 131072;
  unsigned short* wo2f = wo1f + 73728;
  float* agg   = (float*)((char*)d_ws + 540672);
  float* x0    = agg + (size_t)2048 * 256;
  float* xeval = x0 + 8192;
  float* accum = xeval + 8192;

  pack_w2_kernel <<<512, 256, 0, stream>>>(W2, w2f);
  pack_wo1_kernel<<<288, 256, 0, stream>>>(Wo1, wo1f);
  pack_wo2_kernel<<<256, 256, 0, stream>>>(Wo2, wo2f);
  init_x0_kernel <<<32, 256, 0, stream>>>(inputs, x0);

  for (int s = 0; s < 10; ++s) {
    for (int stage = 0; stage < 4; ++stage) {
      const float* xin = (stage == 0) ? x0 : xeval;
      edge_kernel<<<2048, 256, 0, stream>>>(xin, edges, W1, b1, b2, w2f, agg);
      node_kernel<<<64, 256, 0, stream>>>(xin, agg, wo1f, bo1, wo2f, bo2,
                                          Wo3, bo3, x0, xeval, accum, out,
                                          ts, scale, stage, s);
    }
  }
}
